// OPTDecoderLayer_2130303779378
// MI455X (gfx1250) — matrix operand order repaired and checked by reading, not compiled, not run on hardware
//
#include <hip/hip_runtime.h>

// ---------------------------------------------------------------------------
// Types for CDNA5 WMMA (wave32). 16x16x32 bf16 -> f32 accumulate.
// ---------------------------------------------------------------------------
typedef __attribute__((ext_vector_type(16))) __bf16 v16bf;
typedef __attribute__((ext_vector_type(8)))  float  v8f;

#define WMMA_BF16(A, B, C) \
  __builtin_amdgcn_wmma_f32_16x16x32_bf16(false, (A), false, (B), (short)0, (C), false, false)

static constexpr long MROWS = 4096;   // B*T
static constexpr long CDIM  = 2048;
static constexpr long FFDIM = 8192;
static constexpr int  TLEN  = 2048;
static constexpr int  NHEAD = 32;
static constexpr int  DHEAD = 64;

// ---------------------------------------------------------------------------
// CDNA5 async global->LDS copy (ASYNCcnt-tracked, ISA 10.x / 15.18).
// Each active lane copies 16 bytes from its global address to its LDS offset.
// ---------------------------------------------------------------------------
__device__ __forceinline__ void async_copy16(unsigned lds_off, const void* gptr) {
  asm volatile("global_load_async_to_lds_b128 %0, %1, off"
               :
               : "v"(lds_off), "v"((unsigned long long)(size_t)gptr)
               : "memory");
}
__device__ __forceinline__ void wait_async0() {
  asm volatile("s_wait_asynccnt 0" ::: "memory");
}

// ---------------------------------------------------------------------------
// fp32 -> bf16 conversion (vectorized, HBM bound)
// ---------------------------------------------------------------------------
__global__ __launch_bounds__(256) void cvt_bf16_kernel(const float* __restrict__ in,
                                                       __bf16* __restrict__ out, long n) {
  long i = ((long)blockIdx.x * 256 + threadIdx.x) * 4;
  if (i >= n) return;
  float4 v = *(const float4*)(in + i);
  alignas(8) __bf16 o[4] = {(__bf16)v.x, (__bf16)v.y, (__bf16)v.z, (__bf16)v.w};
  *(uint2*)(out + i) = *(const uint2*)o;
}

__device__ __forceinline__ float gelu_exact(float x) {
  return 0.5f * x * (1.0f + erff(x * 0.70710678118654752f));
}

// ---------------------------------------------------------------------------
// Tiled bf16 GEMM: C[M,N] = A[M,K] @ B[K,N] + bias, optional exact GELU.
// Block tile 128x128, 256 threads (8 waves, 4x2 wave grid), K-step 32.
// Each wave owns 32x64: 2 A-frags x 4 B-frags -> 8 WMMAs / K-step, B reused.
// A tile: async global->LDS (double-buffered), row-major [m][k] (K-contiguous).
// B tile: staged through VGPRs with a (k,k+1)-pair pack and stored TRANSPOSED
//         as Bt[n][k] so B-fragment reads are K-contiguous ds_load_b128.
// ---------------------------------------------------------------------------
template <bool GELU>
__global__ __launch_bounds__(256) void gemm_bf16_kernel(
    const __bf16* __restrict__ A, const __bf16* __restrict__ Bm,
    const float* __restrict__ bias, float* __restrict__ Cf,
    __bf16* __restrict__ Cb, int Mdim, int Ndim, int Kdim) {
  __shared__ __bf16 As[2][128][40];   // [m][k]: 128 rows x 32 K (+8 pad), stride 80B
  __shared__ __bf16 Bt[2][128][40];   // [n][k]: 128 cols x 32 K (+8 pad), stride 80B

  const int tid  = threadIdx.x;
  const int wave = tid >> 5;
  const int lane = tid & 31;
  const long bm = (long)blockIdx.y * 128;
  const long bn = (long)blockIdx.x * 128;
  const int mw = wave >> 1, nw = wave & 1;   // 4x2 wave grid
  const int m0 = mw * 32, n0 = nw * 64;

  v8f acc[2][4] = {};

  // A loader: 1 row, 16 elems (2 x b128 async)
  const int ar = tid >> 1, ac = (tid & 1) * 16;
  // B loader: (k,k+1) pair x 8 n
  const int bk = (tid >> 4) * 2;       // 0..30
  const int bn8 = (tid & 15) * 8;      // 0..120

  auto stageA = [&](int k0, int buf) {
#pragma unroll
    for (int j = 0; j < 2; j++)
      async_copy16((unsigned)(size_t)&As[buf][ar][ac + j * 8],
                   A + (bm + ar) * Kdim + k0 + ac + j * 8);
  };

  alignas(16) __bf16 bst[2][8];        // staged B rows k, k+1 (8 n each)
  auto loadB = [&](int k0) {
    *(uint4*)&bst[0][0] = *(const uint4*)(Bm + (long)(k0 + bk) * Ndim + bn + bn8);
    *(uint4*)&bst[1][0] = *(const uint4*)(Bm + (long)(k0 + bk + 1) * Ndim + bn + bn8);
  };
  auto storeB = [&](int buf) {
#pragma unroll
    for (int j = 0; j < 8; j++) {
      alignas(4) __bf16 pr[2] = {bst[0][j], bst[1][j]};
      *(unsigned*)&Bt[buf][bn8 + j][bk] = *(const unsigned*)pr;
    }
  };

  stageA(0, 0);
  loadB(0);
  storeB(0);
  wait_async0();
  __syncthreads();

  const int nsteps = Kdim >> 5;
  const int kh = (lane >> 4) * 8;
  for (int s = 0; s < nsteps; s++) {
    const int buf = s & 1;
    if (s + 1 < nsteps) {
      stageA((s + 1) * 32, buf ^ 1);   // async prefetch next A tile
      loadB((s + 1) * 32);             // global loads in flight during compute
    }

    // A fragments (16x32): 2 x 16B runs per lane -> ds_load_b128 pairs
    v16bf a[2];
#pragma unroll
    for (int mi = 0; mi < 2; mi++) {
      const int arow = m0 + mi * 16 + (lane & 15);
#pragma unroll
      for (int i = 0; i < 8; i++) {
        const int kb = (lane >> 4) * 16 + i * 2;
        a[mi][2 * i]     = As[buf][arow][kb];
        a[mi][2 * i + 1] = As[buf][arow][kb + 1];
      }
    }
#pragma unroll
    for (int nt = 0; nt < 4; nt++) {
      // B fragment (32x16): lane = N; K-contiguous from Bt[n][k]
      v16bf b;
      const int n   = n0 + nt * 16 + (lane & 15);
      const int kb2 = (lane >> 4) * 16;
#pragma unroll
      for (int i = 0; i < 16; i++) b[i] = Bt[buf][n][kb2 + i];
      acc[0][nt] = WMMA_BF16(a[0], b, acc[0][nt]);
      acc[1][nt] = WMMA_BF16(a[1], b, acc[1][nt]);
    }
    if (s + 1 < nsteps) storeB(buf ^ 1);  // transpose-pack B into other buffer
    wait_async0();     // own A prefetch landed in LDS
    __syncthreads();   // all reads of buf done; B stores visible
  }

  // C/D layout: VGPR r -> M = r (lanes 0-15) or r+8 (lanes 16-31); N = lane&15
#pragma unroll
  for (int mi = 0; mi < 2; mi++) {
#pragma unroll
    for (int nt = 0; nt < 4; nt++) {
      const long col = bn + n0 + nt * 16 + (lane & 15);
      const float bv = bias ? bias[col] : 0.0f;
#pragma unroll
      for (int r = 0; r < 8; r++) {
        const long rowg = bm + m0 + mi * 16 + r + ((lane >> 4) * 8);
        float v = acc[mi][nt][r] + bv;
        if (GELU) v = gelu_exact(v);
        const long idx = rowg * Ndim + col;
        if (Cf) Cf[idx] = v;
        if (Cb) Cb[idx] = (__bf16)v;
      }
    }
  }
}

// ---------------------------------------------------------------------------
// Causal flash attention, bf16 WMMA, one (b,h) column of 64 queries per block.
// 128 threads = 4 waves; each wave owns 16 query rows. DH = 64.
// - 1/sqrt(DH) folded into the Q fragments.
// - Diagonal k-tile peeled: main loop carries no mask work.
// - K tiles: async global->LDS double-buffered, [key][dh] (K-contig for QK^T).
// - V tiles: staged through VGPRs, (key,key+1)-pair packed, stored TRANSPOSED
//   as Vt[dh][key] so P@V B-fragments are K-contiguous ds_load_b128.
// ---------------------------------------------------------------------------
__global__ __launch_bounds__(128) void flash_attn_kernel(
    const __bf16* __restrict__ Q, const __bf16* __restrict__ K,
    const __bf16* __restrict__ V, __bf16* __restrict__ O) {
  __shared__ __bf16 Qs[64][72];
  __shared__ __bf16 Ks[2][64][72];   // [key][dh]
  __shared__ __bf16 Vt[2][64][72];   // [dh][key]  (transposed V)
  __shared__ __bf16 Ps[4][16][72];   // per-wave P tile (C-layout -> A-layout relay)

  const int qt   = blockIdx.x;          // query tile (T/64)
  const int bh   = blockIdx.y;          // b * H + h
  const int bb   = bh >> 5;             // H == 32
  const int hh   = bh & 31;
  const int tid  = threadIdx.x;
  const int wave = tid >> 5;
  const int lane = tid & 31;
  const int q0   = qt * 64;

  const int lr = tid >> 1;              // async loader: row 0..63
  const int lc = (tid & 1) * 32;        // col 0 or 32
  const int vkey = (tid >> 2) * 2;      // V loader: key pair 0..62
  const int vdh0 = (tid & 3) * 16;      // V loader: dh group

  auto stageK = [&](int kt, int buf) {
    const long rb = (long)(bb * TLEN + kt * 64 + lr) * CDIM + hh * DHEAD;
#pragma unroll
    for (int j = 0; j < 4; j++)
      async_copy16((unsigned)(size_t)&Ks[buf][lr][lc + j * 8], K + rb + lc + j * 8);
  };

  alignas(16) __bf16 vst[2][16];        // staged V rows key, key+1 (16 dh each)
  auto loadV = [&](int kt) {
    const long rb = (long)(bb * TLEN + kt * 64 + vkey) * CDIM + hh * DHEAD + vdh0;
    *(uint4*)&vst[0][0] = *(const uint4*)(V + rb);
    *(uint4*)&vst[0][8] = *(const uint4*)(V + rb + 8);
    *(uint4*)&vst[1][0] = *(const uint4*)(V + rb + CDIM);
    *(uint4*)&vst[1][8] = *(const uint4*)(V + rb + CDIM + 8);
  };
  auto storeV = [&](int buf) {
#pragma unroll
    for (int j = 0; j < 16; j++) {
      alignas(4) __bf16 pr[2] = {vst[0][j], vst[1][j]};
      *(unsigned*)&Vt[buf][vdh0 + j][vkey] = *(const unsigned*)pr;
    }
  };

  // stage Q tile + first K tile (async) + first V tile (transpose path)
  {
    const long rb = (long)(bb * TLEN + q0 + lr) * CDIM + hh * DHEAD;
#pragma unroll
    for (int j = 0; j < 4; j++)
      async_copy16((unsigned)(size_t)&Qs[lr][lc + j * 8], Q + rb + lc + j * 8);
  }
  stageK(0, 0);
  loadV(0);
  storeV(0);
  wait_async0();
  __syncthreads();

  // Q fragments for K-steps 0 and 1 (dh 0..31, 32..63), pre-scaled by 1/sqrt(64)
  const int arow = wave * 16 + (lane & 15);
  const int kh   = (lane >> 4) * 8;
  v16bf qf[2];
#pragma unroll
  for (int s = 0; s < 2; s++) {
#pragma unroll
    for (int i = 0; i < 8; i++) {
      const int kb = s * 32 + (lane >> 4) * 16 + i * 2;
      qf[s][2 * i]     = (__bf16)((float)Qs[arow][kb] * 0.125f);
      qf[s][2 * i + 1] = (__bf16)((float)Qs[arow][kb + 1] * 0.125f);
    }
  }

  float m_s[8], l_s[8];
#pragma unroll
  for (int r = 0; r < 8; r++) { m_s[r] = -1e30f; l_s[r] = 0.0f; }
  v8f o_acc[4] = {};

  const int qrow_base = q0 + wave * 16 + ((lane >> 4) * 8);
  const int prow = lane & 15;

  auto tile = [&](int kt, bool masked) {
    const int buf   = kt & 1;
    const int kbase = kt * 64;
    if (kt < qt) {
      stageK(kt + 1, buf ^ 1);  // async prefetch next K tile
      loadV(kt + 1);            // next V in flight during compute
    }

    // S = Q @ K^T : B operand is K^T, so b[k=dh][n=key] = Ks[key][dh] (K-contig)
    v8f s_acc[4] = {};
#pragma unroll
    for (int ks = 0; ks < 2; ks++) {
#pragma unroll
      for (int nt = 0; nt < 4; nt++) {
        v16bf bfr;
        const int n   = nt * 16 + (lane & 15);
        const int kb2 = ks * 32 + (lane >> 4) * 16;
#pragma unroll
        for (int i = 0; i < 16; i++) bfr[i] = Ks[buf][n][kb2 + i];
        s_acc[nt] = WMMA_BF16(qf[ks], bfr, s_acc[nt]);
      }
    }

    // mask only on the (peeled) diagonal tile
    if (masked) {
#pragma unroll
      for (int nt = 0; nt < 4; nt++) {
        const int kg = kbase + nt * 16 + (lane & 15);
#pragma unroll
        for (int r = 0; r < 8; r++)
          if (kg > qrow_base + r) s_acc[nt][r] = -1e30f;
      }
    }

    // online softmax: row max / exp / row sum; rows live across one 16-lane half
    float alpha[8];
#pragma unroll
    for (int r = 0; r < 8; r++) {
      float mx = fmaxf(fmaxf(s_acc[0][r], s_acc[1][r]), fmaxf(s_acc[2][r], s_acc[3][r]));
#pragma unroll
      for (int off = 1; off < 16; off <<= 1) mx = fmaxf(mx, __shfl_xor(mx, off, 32));
      const float mnew = fmaxf(m_s[r], mx);
      alpha[r] = __expf(m_s[r] - mnew);
      m_s[r]   = mnew;
    }
#pragma unroll
    for (int r = 0; r < 8; r++) {
      float rs = 0.0f;
#pragma unroll
      for (int nt = 0; nt < 4; nt++) {
        const float p = __expf(s_acc[nt][r] - m_s[r]);
        rs += p;
        // relay P to LDS row-major so it can be refetched in A-fragment layout
        Ps[wave][r + ((lane >> 4) * 8)][nt * 16 + (lane & 15)] = (__bf16)p;
      }
#pragma unroll
      for (int off = 1; off < 16; off <<= 1) rs += __shfl_xor(rs, off, 32);
      l_s[r] = l_s[r] * alpha[r] + rs;
#pragma unroll
      for (int nt = 0; nt < 4; nt++) o_acc[nt][r] *= alpha[r];
    }

    // O += P @ V : B operand from Vt[dh][key] (K-contig per lane)
#pragma unroll
    for (int ks = 0; ks < 2; ks++) {
      v16bf pf;
#pragma unroll
      for (int i = 0; i < 8; i++) {
        const int kb = ks * 32 + (lane >> 4) * 16 + i * 2;
        pf[2 * i]     = Ps[wave][prow][kb];
        pf[2 * i + 1] = Ps[wave][prow][kb + 1];
      }
#pragma unroll
      for (int nt = 0; nt < 4; nt++) {
        v16bf vfr;
        const int n   = nt * 16 + (lane & 15);
        const int kb2 = ks * 32 + (lane >> 4) * 16;
#pragma unroll
        for (int i = 0; i < 16; i++) vfr[i] = Vt[buf][n][kb2 + i];
        o_acc[nt] = WMMA_BF16(pf, vfr, o_acc[nt]);
      }
    }
    if (kt < qt) {
      storeV(buf ^ 1); // transpose-pack next V into the other buffer
      wait_async0();   // next K tile landed
      __syncthreads(); // everyone done reading buf; V stores visible
    }
  };

  for (int kt = 0; kt < qt; kt++) tile(kt, false);  // mask-free main loop
  tile(qt, true);                                   // peeled diagonal tile

  // epilogue: O /= l, store bf16 at [b, t, h, dh] (== flat [B*T, C])
#pragma unroll
  for (int nt = 0; nt < 4; nt++) {
    const int dh = nt * 16 + (lane & 15);
#pragma unroll
    for (int r = 0; r < 8; r++) {
      const int  trow = q0 + wave * 16 + r + ((lane >> 4) * 8);
      const float inv = 1.0f / l_s[r];
      O[((long)(bb * TLEN + trow)) * CDIM + hh * DHEAD + dh] = (__bf16)(o_acc[nt][r] * inv);
    }
  }
}

// ---------------------------------------------------------------------------
// Fused residual-add + LayerNorm over rows of 2048. One block per row.
// Produces f32 (residual chain / final output) and optional bf16 copy.
// ---------------------------------------------------------------------------
__global__ __launch_bounds__(256) void add_ln_kernel(
    const float* __restrict__ a, const float* __restrict__ res,
    const float* __restrict__ g, const float* __restrict__ be,
    float* __restrict__ outf, __bf16* __restrict__ outb) {
  const int row = blockIdx.x;
  const int tid = threadIdx.x;
  const float* pa = a + (long)row * CDIM;
  const float* pr = res + (long)row * CDIM;

  __shared__ float sm[8];
  __shared__ float mean_sh, rstd_sh;

  float v[8];
  float s = 0.0f;
#pragma unroll
  for (int i = 0; i < 8; i++) {
    const int c = tid + i * 256;
    v[i] = pa[c] + pr[c];
    s += v[i];
  }
#pragma unroll
  for (int off = 16; off; off >>= 1) s += __shfl_xor(s, off, 32);
  if ((tid & 31) == 0) sm[tid >> 5] = s;
  __syncthreads();
  if (tid == 0) {
    float t = 0.0f;
    for (int i = 0; i < 8; i++) t += sm[i];
    mean_sh = t * (1.0f / 2048.0f);
  }
  __syncthreads();
  const float mean = mean_sh;

  float vs = 0.0f;
#pragma unroll
  for (int i = 0; i < 8; i++) {
    const float d = v[i] - mean;
    vs += d * d;
  }
#pragma unroll
  for (int off = 16; off; off >>= 1) vs += __shfl_xor(vs, off, 32);
  if ((tid & 31) == 0) sm[tid >> 5] = vs;
  __syncthreads();
  if (tid == 0) {
    float t = 0.0f;
    for (int i = 0; i < 8; i++) t += sm[i];
    rstd_sh = rsqrtf(t * (1.0f / 2048.0f) + 1e-5f);
  }
  __syncthreads();
  const float rstd = rstd_sh;

#pragma unroll
  for (int i = 0; i < 8; i++) {
    const int c = tid + i * 256;
    const float o = (v[i] - mean) * rstd * g[c] + be[c];
    outf[(long)row * CDIM + c] = o;
    if (outb) outb[(long)row * CDIM + c] = (__bf16)o;
  }
}

// ---------------------------------------------------------------------------
// Orchestration
// ---------------------------------------------------------------------------
extern "C" void kernel_launch(void* const* d_in, const int* in_sizes, int n_in,
                              void* d_out, int out_size, void* d_ws, size_t ws_size,
                              hipStream_t stream) {
  const float* x   = (const float*)d_in[0];
  const float* Wq  = (const float*)d_in[1];
  const float* bq  = (const float*)d_in[2];
  const float* Wk  = (const float*)d_in[3];
  const float* bk  = (const float*)d_in[4];
  const float* Wv  = (const float*)d_in[5];
  const float* bv  = (const float*)d_in[6];
  const float* Wo  = (const float*)d_in[7];
  const float* bo  = (const float*)d_in[8];
  const float* g1  = (const float*)d_in[9];
  const float* b1  = (const float*)d_in[10];
  const float* W1  = (const float*)d_in[11];
  const float* bf1 = (const float*)d_in[12];
  const float* W2  = (const float*)d_in[13];
  const float* bf2 = (const float*)d_in[14];
  const float* g2  = (const float*)d_in[15];
  const float* b2  = (const float*)d_in[16];
  float* out = (float*)d_out;

  char* ws = (char*)d_ws;
  size_t off = 0;
  auto walloc = [&](size_t bytes) -> void* {
    void* p = ws + off;
    off += (bytes + 255) & ~(size_t)255;
    return p;
  };

  __bf16* xb   = (__bf16*)walloc(MROWS * CDIM * 2);
  __bf16* wqb  = (__bf16*)walloc(CDIM * CDIM * 2);
  __bf16* wkb  = (__bf16*)walloc(CDIM * CDIM * 2);
  __bf16* wvb  = (__bf16*)walloc(CDIM * CDIM * 2);
  __bf16* wob  = (__bf16*)walloc(CDIM * CDIM * 2);
  __bf16* w1b  = (__bf16*)walloc(CDIM * FFDIM * 2);
  __bf16* w2b  = (__bf16*)walloc(FFDIM * CDIM * 2);
  __bf16* qb   = (__bf16*)walloc(MROWS * CDIM * 2);
  __bf16* kb   = (__bf16*)walloc(MROWS * CDIM * 2);
  __bf16* vb   = (__bf16*)walloc(MROWS * CDIM * 2);
  __bf16* attb = (__bf16*)walloc(MROWS * CDIM * 2);
  float*  proj = (float*)walloc(MROWS * CDIM * 4);
  float*  hf   = (float*)walloc(MROWS * CDIM * 4);
  __bf16* hb   = (__bf16*)walloc(MROWS * CDIM * 2);
  __bf16* f1b  = (__bf16*)walloc(MROWS * FFDIM * 2);
  float*  ff   = (float*)walloc(MROWS * CDIM * 4);

  auto cvt = [&](const float* src, __bf16* dst, long n) {
    const long nt4 = n / 4;
    cvt_bf16_kernel<<<(int)((nt4 + 255) / 256), 256, 0, stream>>>(src, dst, n);
  };

  // 1) precision conversion
  cvt(x, xb, MROWS * CDIM);
  cvt(Wq, wqb, CDIM * CDIM);
  cvt(Wk, wkb, CDIM * CDIM);
  cvt(Wv, wvb, CDIM * CDIM);
  cvt(Wo, wob, CDIM * CDIM);
  cvt(W1, w1b, CDIM * FFDIM);
  cvt(W2, w2b, FFDIM * CDIM);

  const dim3 blk(256);
  // 2) Q/K/V projections (bf16 out for attention)
  {
    dim3 grid((unsigned)(CDIM / 128), (unsigned)(MROWS / 128));
    gemm_bf16_kernel<false><<<grid, blk, 0, stream>>>(xb, wqb, bq, nullptr, qb,
                                                      (int)MROWS, (int)CDIM, (int)CDIM);
    gemm_bf16_kernel<false><<<grid, blk, 0, stream>>>(xb, wkb, bk, nullptr, kb,
                                                      (int)MROWS, (int)CDIM, (int)CDIM);
    gemm_bf16_kernel<false><<<grid, blk, 0, stream>>>(xb, wvb, bv, nullptr, vb,
                                                      (int)MROWS, (int)CDIM, (int)CDIM);
  }

  // 3) causal flash attention
  {
    dim3 grid((unsigned)(TLEN / 64), (unsigned)(2 * NHEAD));
    flash_attn_kernel<<<grid, dim3(128), 0, stream>>>(qb, kb, vb, attb);
  }

  // 4) output projection (f32 out)
  {
    dim3 grid((unsigned)(CDIM / 128), (unsigned)(MROWS / 128));
    gemm_bf16_kernel<false><<<grid, blk, 0, stream>>>(attb, wob, bo, proj, nullptr,
                                                      (int)MROWS, (int)CDIM, (int)CDIM);
  }

  // 5) h = LN1(proj + x)  -> f32 residual + bf16 GEMM input
  add_ln_kernel<<<(unsigned)MROWS, 256, 0, stream>>>(proj, x, g1, b1, hf, hb);

  // 6) FFN up + exact GELU (bf16 out)
  {
    dim3 grid((unsigned)(FFDIM / 128), (unsigned)(MROWS / 128));
    gemm_bf16_kernel<true><<<grid, blk, 0, stream>>>(hb, w1b, bf1, nullptr, f1b,
                                                     (int)MROWS, (int)FFDIM, (int)CDIM);
  }

  // 7) FFN down (f32 out)
  {
    dim3 grid((unsigned)(CDIM / 128), (unsigned)(MROWS / 128));
    gemm_bf16_kernel<false><<<grid, blk, 0, stream>>>(f1b, w2b, bf2, ff, nullptr,
                                                      (int)MROWS, (int)CDIM, (int)FFDIM);
  }

  // 8) out = LN2(ff + h)
  add_ln_kernel<<<(unsigned)MROWS, 256, 0, stream>>>(ff, hf, g2, b2, out, nullptr);
}
